// Seq2Seq_44341242364275
// MI455X (gfx1250) — matrix operand order repaired and checked by reading, not compiled, not run on hardware
//
#include <hip/hip_runtime.h>
#include <stddef.h>

// ---------------------------------------------------------------------------
// Seq2Seq LSTM (enc 2-layer -> dec 2-layer teacher forced -> FC) for gfx1250.
//   * batch-partitioned persistent recurrent kernels (16 batch rows / WG,
//     16 WGs, 512 threads = 16 waves), h/c state resident in LDS
//   * gates = [x_t | h_{t-1}] @ [Wih;Whh]^T via v_wmma_f32_16x16x32_bf16,
//     accumulator chain starts from inline-0 C; bias folded into the
//     nonlinearity phase from per-thread registers (j = tid&127 invariant)
//   * weights pre-packed bf16 in per-wave WMMA fragment order (coalesced,
//     L2-resident); A operands gathered from LDS per documented layout
//   * next-timestep input prefetch (global_prefetch_b8)
//   * decoder layer-1 hidden sequence stored bf16 -> big WMMA GEMM for FC
// ---------------------------------------------------------------------------

typedef __attribute__((ext_vector_type(16))) __bf16 v16bf;
typedef __attribute__((ext_vector_type(8)))  float  v8f;
typedef __attribute__((ext_vector_type(4)))  unsigned int uint4e;

typedef unsigned short bf16_t;   // raw bf16 storage

union FragU { v16bf v; uint4e q[2]; };

__device__ __forceinline__ bf16_t f2bf(float f) {
  union { float f; unsigned int u; } v; v.f = f;
  unsigned int r = v.u + 0x7FFFu + ((v.u >> 16) & 1u);   // round-to-nearest-even
  return (bf16_t)(r >> 16);
}

// A fragment: 16-bit A 16x32 layout. Caller pre-offsets pointer by
// (lane&15)*rowStride + (lane>>4)*8.  Chunks at +0 and +16 elements.
__device__ __forceinline__ v16bf load_fragA(const bf16_t* p) {
  FragU u;
  u.q[0] = *reinterpret_cast<const uint4e*>(p);
  u.q[1] = *reinterpret_cast<const uint4e*>(p + 16);
  return u.v;
}
// B fragment (pre-tiled): 16 contiguous bf16 per lane -> chunks +0, +8.
__device__ __forceinline__ v16bf load_frag32(const bf16_t* p) {
  FragU u;
  u.q[0] = *reinterpret_cast<const uint4e*>(p);
  u.q[1] = *reinterpret_cast<const uint4e*>(p + 8);
  return u.v;
}

// ---------------------------------------------------------------------------
// Weight packing: Wt fragment order [ntile(32)][kstep(KTOT/32)][lane(32)][16],
// lane<16: col n0+lane, k = k0+0..7,16..23 ; lane>=16: col n0+lane-16, k = k0+8..15,24..31
// ---------------------------------------------------------------------------
__global__ void pack_wcat_kernel(const float* __restrict__ Wih,
                                 const float* __restrict__ Whh,
                                 const float* __restrict__ bih,
                                 const float* __restrict__ bhh,
                                 bf16_t* __restrict__ Wt,
                                 float* __restrict__ bias,
                                 int Kin, int Kinpad)
{
  const int ktot  = Kinpad + 128;
  const int KS    = ktot >> 5;
  const int total = 512 * ktot;
  for (int idx = blockIdx.x * blockDim.x + threadIdx.x; idx < total;
       idx += gridDim.x * blockDim.x) {
    const int e    = idx & 15;
    const int lane = (idx >> 4) & 31;
    const int frag = idx >> 9;          // = ntile*KS + ks
    const int ks   = frag % KS;
    const int nt   = frag / KS;
    const int n    = nt * 16 + (lane & 15);
    const int k    = ks * 32 + (lane >> 4) * 8 + (e & 7) + (e >> 3) * 16;
    float v;
    if (k < Kinpad) v = (k < Kin) ? Wih[n * Kin + k] : 0.f;
    else            v = Whh[n * 128 + (k - Kinpad)];
    Wt[idx] = f2bf(v);
  }
  const int gi = blockIdx.x * blockDim.x + threadIdx.x;
  if (gi < 512) bias[gi] = bih[gi] + bhh[gi];
}

__global__ void pack_fc_kernel(const float* __restrict__ W,
                               const float* __restrict__ b,
                               bf16_t* __restrict__ Wt,
                               float* __restrict__ bias)
{
  const int total = 13 * 4 * 32 * 16;   // 208 cols x 128 k, fragment order
  for (int idx = blockIdx.x * blockDim.x + threadIdx.x; idx < total;
       idx += gridDim.x * blockDim.x) {
    const int e    = idx & 15;
    const int lane = (idx >> 4) & 31;
    const int frag = idx >> 9;          // nt*4 + ks
    const int ks   = frag & 3;
    const int nt   = frag >> 2;
    const int n    = nt * 16 + (lane & 15);
    const int k    = ks * 32 + (lane >> 4) * 8 + (e & 7) + (e >> 3) * 16;
    Wt[idx] = f2bf((n < 200) ? W[n * 128 + k] : 0.f);
  }
  const int gi = blockIdx.x * blockDim.x + threadIdx.x;
  if (gi < 208) bias[gi] = (gi < 200) ? b[gi] : 0.f;
}

// ---------------------------------------------------------------------------
// Per-wave gate GEMM: wave owns gate columns [wave*32, wave*32+32), i.e. two
// 16x16 N-tiles. A = [xbuf | hbuf] (LDS), B = pre-tiled weights (global/L2).
// Accumulators start from inline-0 C (bias is applied in the update phase).
// ---------------------------------------------------------------------------
template <int KTOT, int KINP, int XSTRIDE>
__device__ __forceinline__ void gemm_gates(const bf16_t* xbuf,
                                           const bf16_t* hbuf,
                                           const bf16_t* __restrict__ Wt,
                                           float* gbuf, int wave, int lane)
{
  constexpr int KS = KTOT / 32;
  const int mrow = lane & 15;
  const int half = lane >> 4;
  const bf16_t* xrow = xbuf + mrow * XSTRIDE + half * 8;
  const bf16_t* hrow = hbuf + mrow * 136 + half * 8;
#pragma unroll
  for (int nt = 0; nt < 2; ++nt) {
    const int ntg  = wave * 2 + nt;
    const int ncol = ntg * 16 + mrow;
    v8f acc = {};                       // -> v_wmma ..., 0 (inline zero C)
    const bf16_t* bfrag = Wt + ((size_t)(ntg * KS) * 32 + lane) * 16;
#pragma unroll
    for (int ks = 0; ks < KS; ++ks) {
      const int k0 = ks * 32;
      const bf16_t* ar = (k0 < KINP) ? (xrow + k0) : (hrow + (k0 - KINP));
      v16bf a = load_fragA(ar);
      v16bf b = load_frag32(bfrag);
      acc = __builtin_amdgcn_wmma_f32_16x16x32_bf16(
          false, a, false, b, (short)0, acc, false, false);
      bfrag += 512;   // next k-step fragment (32 lanes * 16 elems)
    }
    const int mb = half * 8;   // C/D layout: VGPR r -> row mb+r, col ncol
#pragma unroll
    for (int r = 0; r < 8; ++r)
      gbuf[(mb + r) * 514 + ncol] = acc[r];
  }
}

// ---------------------------------------------------------------------------
// Fused 2-layer LSTM over the full sequence. One WG owns 16 batch rows.
// ---------------------------------------------------------------------------
template <int KIN, int KINPAD, bool IS_DEC>
__global__ __launch_bounds__(512)
void lstm2_kernel(const float* __restrict__ in_seq,              // [B][1024][KIN]
                  const bf16_t* __restrict__ Wt0, const float* __restrict__ bias0,
                  const bf16_t* __restrict__ Wt1, const float* __restrict__ bias1,
                  const bf16_t* __restrict__ h_init,             // [2][B][128] (dec)
                  const float*  __restrict__ c_init,             // [2][B][128] (dec)
                  bf16_t* __restrict__ h_final,                  // [2][B][128] (enc)
                  float*  __restrict__ c_final,                  // [2][B][128] (enc)
                  bf16_t* __restrict__ out_seq)                  // [B][1024][128] (dec)
{
  constexpr int KT0  = KINPAD + 128;
  constexpr int XSTR = (KINPAD == 128) ? 136 : KINPAD;  // 16B-aligned rows

  __shared__ bf16_t xbuf[16 * XSTR];
  __shared__ bf16_t h0buf[16 * 136];
  __shared__ bf16_t h1buf[16 * 136];
  __shared__ float  gates[16 * 514];
  __shared__ float  cbuf0[16 * 130];
  __shared__ float  cbuf1[16 * 130];

  const int tid  = threadIdx.x;
  const int wave = tid >> 5;
  const int lane = tid & 31;
  const int b0   = blockIdx.x * 16;

  // Per-thread bias registers: gate column j = tid&127 is invariant across
  // the 4 update elements and across all timesteps.
  const int jj = tid & 127;
  const float b0i = bias0[jj],       b0f = bias0[128 + jj];
  const float b0g = bias0[256 + jj], b0o = bias0[384 + jj];
  const float b1i = bias1[jj],       b1f = bias1[128 + jj];
  const float b1g = bias1[256 + jj], b1o = bias1[384 + jj];

  // ---- init state (zero for encoder, enc-final for decoder) ----
  for (int idx = tid; idx < 16 * 136; idx += 512) {
    const int m = idx / 136, j = idx % 136;
    bf16_t h0v = 0, h1v = 0;
    float  c0v = 0.f, c1v = 0.f;
    if (IS_DEC && j < 128) {
      h0v = h_init[(size_t)(b0 + m) * 128 + j];
      h1v = h_init[(size_t)(256 + b0 + m) * 128 + j];
      c0v = c_init[(size_t)(b0 + m) * 128 + j];
      c1v = c_init[(size_t)(256 + b0 + m) * 128 + j];
    }
    h0buf[idx] = h0v;
    h1buf[idx] = h1v;
    if (j < 130) { cbuf0[m * 130 + j] = c0v; cbuf1[m * 130 + j] = c1v; }
  }
  for (int idx = tid; idx < 16 * XSTR; idx += 512) {
    if ((idx % XSTR) >= KIN) xbuf[idx] = 0;   // zero K padding once
  }
  __syncthreads();

  for (int t = 0; t < 1024; ++t) {
    // ---- stage layer-0 input tile (decoder: shifted teacher forcing) ----
    if (!IS_DEC || t > 0) {
      const int toff = IS_DEC ? (t - 1) : t;
      for (int idx = tid; idx < 16 * KIN; idx += 512) {
        const int m = idx / KIN, i = idx - m * KIN;
        const float* src = &in_seq[((size_t)(b0 + m) * 1024 + toff) * KIN + i];
        xbuf[m * XSTR + i] = f2bf(*src);
        if (toff + 1 < 1024)
          __builtin_prefetch(src + KIN, 0, 0);   // global_prefetch_b8 (t+1)
      }
    } else {
      for (int idx = tid; idx < 16 * KIN; idx += 512) {
        const int m = idx / KIN, i = idx - m * KIN;
        xbuf[m * XSTR + i] = 0;
      }
    }
    __syncthreads();

    // ---- layer 0: gates = [x_t | h0] @ W0^T ----
    gemm_gates<KT0, KINPAD, XSTR>(xbuf, h0buf, Wt0, gates, wave, lane);
    __syncthreads();

    // ---- layer 0 state update (PyTorch gate order i,f,g,o) ----
#pragma unroll
    for (int e = 0; e < 4; ++e) {
      const int jl = e * 512 + tid;
      const int m = jl >> 7, j = jl & 127;
      const float iv = gates[m * 514 + j]       + b0i;
      const float fv = gates[m * 514 + 128 + j] + b0f;
      const float gv = gates[m * 514 + 256 + j] + b0g;
      const float ov = gates[m * 514 + 384 + j] + b0o;
      const float is = 1.f / (1.f + __expf(-iv));
      const float fs = 1.f / (1.f + __expf(-fv));
      const float os = 1.f / (1.f + __expf(-ov));
      const float gt = tanhf(gv);
      const float c  = fs * cbuf0[m * 130 + j] + is * gt;
      cbuf0[m * 130 + j] = c;
      h0buf[m * 136 + j] = f2bf(os * tanhf(c));
    }
    __syncthreads();

    // ---- layer 1: gates = [h0_t | h1] @ W1^T ----
    gemm_gates<256, 128, 136>(h0buf, h1buf, Wt1, gates, wave, lane);
    __syncthreads();

    // ---- layer 1 state update (+ optional sequence store) ----
#pragma unroll
    for (int e = 0; e < 4; ++e) {
      const int jl = e * 512 + tid;
      const int m = jl >> 7, j = jl & 127;
      const float iv = gates[m * 514 + j]       + b1i;
      const float fv = gates[m * 514 + 128 + j] + b1f;
      const float gv = gates[m * 514 + 256 + j] + b1g;
      const float ov = gates[m * 514 + 384 + j] + b1o;
      const float is = 1.f / (1.f + __expf(-iv));
      const float fs = 1.f / (1.f + __expf(-fv));
      const float os = 1.f / (1.f + __expf(-ov));
      const float gt = tanhf(gv);
      const float c  = fs * cbuf1[m * 130 + j] + is * gt;
      cbuf1[m * 130 + j] = c;
      const bf16_t hb = f2bf(os * tanhf(c));
      h1buf[m * 136 + j] = hb;
      if (IS_DEC)
        out_seq[((size_t)(b0 + m) * 1024 + t) * 128 + j] = hb;
    }
    __syncthreads();
  }

  // ---- encoder: emit final h/c for both layers ----
  if (!IS_DEC) {
    for (int idx = tid; idx < 16 * 128; idx += 512) {
      const int m = idx >> 7, j = idx & 127;
      h_final[(size_t)(b0 + m) * 128 + j]       = h0buf[m * 136 + j];
      h_final[(size_t)(256 + b0 + m) * 128 + j] = h1buf[m * 136 + j];
      c_final[(size_t)(b0 + m) * 128 + j]       = cbuf0[m * 130 + j];
      c_final[(size_t)(256 + b0 + m) * 128 + j] = cbuf1[m * 130 + j];
    }
  }
}

// ---------------------------------------------------------------------------
// Final FC: out[m, 0..200) = dec1_seq[m, :] @ fc_W^T + fc_b   (M = B*S rows)
// One wave per 16-row M-tile; 13 N-tiles of 16 (200 padded to 208).
// ---------------------------------------------------------------------------
__global__ __launch_bounds__(256)
void fc_wmma_kernel(const bf16_t* __restrict__ seq,   // [B*S][128] bf16
                    const bf16_t* __restrict__ Wt,    // fragment order
                    const float*  __restrict__ bias,  // [208]
                    float* __restrict__ out)          // [B*S][200]
{
  const int lane = threadIdx.x & 31;
  const int wave = threadIdx.x >> 5;
  const int mt   = blockIdx.x * 8 + wave;             // 0..16383
  const int mrow = lane & 15;
  const int half = lane >> 4;
  const bf16_t* arow = seq + ((size_t)mt * 16 + mrow) * 128 + half * 8;
#pragma unroll
  for (int nt = 0; nt < 13; ++nt) {
    const int ncol = nt * 16 + mrow;
    v8f acc = {};                       // inline-0 C; bias added at store
    const bf16_t* bfrag = Wt + ((size_t)(nt * 4) * 32 + lane) * 16;
#pragma unroll
    for (int ks = 0; ks < 4; ++ks) {
      v16bf a = load_fragA(arow + ks * 32);
      v16bf b = load_frag32(bfrag);
      acc = __builtin_amdgcn_wmma_f32_16x16x32_bf16(
          false, a, false, b, (short)0, acc, false, false);
      bfrag += 512;
    }
    if (ncol < 200) {
      const float bv = bias[ncol];
      const int mb = half * 8;
#pragma unroll
      for (int r = 0; r < 8; ++r)
        out[((size_t)mt * 16 + mb + r) * 200 + ncol] = acc[r] + bv;
    }
  }
}

// ---------------------------------------------------------------------------
extern "C" void kernel_launch(void* const* d_in, const int* in_sizes, int n_in,
                              void* d_out, int out_size, void* d_ws, size_t ws_size,
                              hipStream_t stream)
{
  (void)in_sizes; (void)n_in; (void)out_size; (void)ws_size;
  const float* x    = (const float*)d_in[0];
  const float* y    = (const float*)d_in[1];
  const float* eW0i = (const float*)d_in[2];
  const float* eW0h = (const float*)d_in[3];
  const float* eb0i = (const float*)d_in[4];
  const float* eb0h = (const float*)d_in[5];
  const float* eW1i = (const float*)d_in[6];
  const float* eW1h = (const float*)d_in[7];
  const float* eb1i = (const float*)d_in[8];
  const float* eb1h = (const float*)d_in[9];
  const float* dW0i = (const float*)d_in[10];
  const float* dW0h = (const float*)d_in[11];
  const float* db0i = (const float*)d_in[12];
  const float* db0h = (const float*)d_in[13];
  const float* dW1i = (const float*)d_in[14];
  const float* dW1h = (const float*)d_in[15];
  const float* db1i = (const float*)d_in[16];
  const float* db1h = (const float*)d_in[17];
  const float* fcW  = (const float*)d_in[18];
  const float* fcb  = (const float*)d_in[19];

  char*  ws  = (char*)d_ws;
  size_t off = 0;
  auto take = [&](size_t bytes) -> char* {
    char* p = ws + off;
    off += (bytes + 255) & ~(size_t)255;
    return p;
  };
  bf16_t* WtE0 = (bf16_t*)take((size_t)512 * 256 * 2);
  bf16_t* WtE1 = (bf16_t*)take((size_t)512 * 256 * 2);
  bf16_t* WtD0 = (bf16_t*)take((size_t)512 * 352 * 2);
  bf16_t* WtD1 = (bf16_t*)take((size_t)512 * 256 * 2);
  bf16_t* WtFC = (bf16_t*)take((size_t)26624 * 2);
  float*  bE0  = (float*)take(512 * 4);
  float*  bE1  = (float*)take(512 * 4);
  float*  bD0  = (float*)take(512 * 4);
  float*  bD1  = (float*)take(512 * 4);
  float*  bFC  = (float*)take(208 * 4);
  bf16_t* hT   = (bf16_t*)take((size_t)2 * 256 * 128 * 2);
  float*  cT   = (float*)take((size_t)2 * 256 * 128 * 4);
  bf16_t* seq  = (bf16_t*)take((size_t)256 * 1024 * 128 * 2);

  // ---- pack weights (bf16, WMMA fragment order) + biases ----
  pack_wcat_kernel<<<512, 256, 0, stream>>>(eW0i, eW0h, eb0i, eb0h, WtE0, bE0, 100, 128);
  pack_wcat_kernel<<<512, 256, 0, stream>>>(eW1i, eW1h, eb1i, eb1h, WtE1, bE1, 128, 128);
  pack_wcat_kernel<<<512, 256, 0, stream>>>(dW0i, dW0h, db0i, db0h, WtD0, bD0, 200, 224);
  pack_wcat_kernel<<<512, 256, 0, stream>>>(dW1i, dW1h, db1i, db1h, WtD1, bD1, 128, 128);
  pack_fc_kernel<<<104, 256, 0, stream>>>(fcW, fcb, WtFC, bFC);

  // ---- encoder (zero init, emits final h/c) ----
  lstm2_kernel<100, 128, false><<<16, 512, 0, stream>>>(
      x, WtE0, bE0, WtE1, bE1,
      (const bf16_t*)nullptr, (const float*)nullptr, hT, cT, (bf16_t*)nullptr);

  // ---- decoder (init from encoder, emits layer-1 sequence) ----
  lstm2_kernel<200, 224, true><<<16, 512, 0, stream>>>(
      y, WtD0, bD0, WtD1, bD1,
      (const bf16_t*)hT, (const float*)cT, (bf16_t*)nullptr, (float*)nullptr, seq);

  // ---- final FC GEMM ----
  fc_wmma_kernel<<<2048, 256, 0, stream>>>(seq, WtFC, bFC, (float*)d_out);
}
